// RelationModel_8658654069162
// MI455X (gfx1250) — matrix operand order repaired and checked by reading, not compiled, not run on hardware
//
#include <hip/hip_runtime.h>

// ---------------------------------------------------------------------------
// RelationModel on MI455X (gfx1250, wave32, WMMA).
//   G = dendron . axon^T / d is step-invariant (act_axon = s[m]*axon[m,:]):
//   computed ONCE with v_wmma_f32_16x16x32_bf16, stored bf16 (42.5MB, fits the
//   192MB L2). Each of 8 steps fuses scale->clip->bf16 into the A-operand of
//   mt = clip(G*s) @ [identity|att], whose B-operand (iaT) is staged into a
//   double-buffered LDS slab via GLOBAL_LOAD_ASYNC_TO_LDS_B128 (ASYNCcnt) and
//   consumed with ds_load_b128. MLP = two more WMMA kernels with fused
//   bias/relu/residual and the att update + history write.
// ---------------------------------------------------------------------------

typedef __attribute__((ext_vector_type(16))) __bf16 v16bf;
typedef __attribute__((ext_vector_type(8)))  __bf16 v8bf;
typedef __attribute__((ext_vector_type(8)))  float  v8f;
typedef __attribute__((ext_vector_type(4)))  int    v4i;

#define GAS_ __attribute__((address_space(1)))
#define LAS_ __attribute__((address_space(3)))

#define B_  16
#define N_  128
#define FD_ 512
#define E_  128
#define I_  64
#define A_  64
#define H_  256
#define M_  1024
#define L_  8
#define D_  1152   // M_ + N_

#if defined(__has_builtin)
#if __has_builtin(__builtin_amdgcn_global_load_async_to_lds_b128) && \
    __has_builtin(__builtin_amdgcn_s_wait_asynccnt)
#define USE_ASYNC_LDS 1
#endif
#endif
#ifndef USE_ASYNC_LDS
#define USE_ASYNC_LDS 0
#endif

static __device__ __forceinline__ float clip2(float x) {
    return fminf(fmaxf(x, -2.0f), 2.0f);
}

// --------------------------- setup kernels ---------------------------------

__global__ void kConcepts(const float* __restrict__ CEin, const float* __restrict__ CEout,
                          const float* __restrict__ CEid,
                          __bf16* __restrict__ dendron, __bf16* __restrict__ axon,
                          __bf16* __restrict__ iaT) {
    int idx = blockIdx.x * blockDim.x + threadIdx.x;   // over B*M
    if (idx >= B_ * M_) return;
    int b = idx / M_, m = idx % M_;
    const float* ci = CEin  + (size_t)m * E_;
    const float* co = CEout + (size_t)m * E_;
    __bf16* drow = dendron + ((size_t)b * D_ + m) * E_;
    __bf16* arow = axon    + ((size_t)b * D_ + m) * E_;
    for (int e = 0; e < E_; ++e) { drow[e] = (__bf16)ci[e]; arow[e] = (__bf16)co[e]; }
    const float* cid = CEid + (size_t)m * I_;
    for (int i = 0; i < I_; ++i)
        iaT[((size_t)b * (I_ + A_) + i) * D_ + m] = (__bf16)cid[i];
}

__global__ void kProj(const float* __restrict__ scene,
                      const float* __restrict__ WfIn, const float* __restrict__ bfIn,
                      const float* __restrict__ WfOut, const float* __restrict__ bfOut,
                      const float* __restrict__ WfId, const float* __restrict__ bfId,
                      __bf16* __restrict__ dendron, __bf16* __restrict__ axon,
                      __bf16* __restrict__ iaT) {
    int idx = blockIdx.x * blockDim.x + threadIdx.x;   // B*N*(E+E+I)
    if (idx >= B_ * N_ * (2 * E_ + I_)) return;
    int j  = idx % (2 * E_ + I_);
    int bn = idx / (2 * E_ + I_);
    int b = bn / N_, n = bn % N_;
    const float* srow = scene + (size_t)bn * FD_;
    const float* W; const float* bias; int col; int wcols;
    if (j < E_)            { W = WfIn;  bias = bfIn;  col = j;          wcols = E_; }
    else if (j < 2 * E_)   { W = WfOut; bias = bfOut; col = j - E_;     wcols = E_; }
    else                   { W = WfId;  bias = bfId;  col = j - 2 * E_; wcols = I_; }
    float acc = bias[col];
    for (int f = 0; f < FD_; ++f) acc += srow[f] * W[(size_t)f * wcols + col];
    int row = M_ + n;
    if (j < E_)          dendron[((size_t)b * D_ + row) * E_ + col] = (__bf16)acc;
    else if (j < 2 * E_) axon   [((size_t)b * D_ + row) * E_ + col] = (__bf16)acc;
    else                 iaT[((size_t)b * (I_ + A_) + col) * D_ + row] = (__bf16)acc;
}

__global__ void kWT(const float* __restrict__ W1, const float* __restrict__ W2,
                    __bf16* __restrict__ W1T, __bf16* __restrict__ W2T,
                    float* __restrict__ att) {
    int idx = blockIdx.x * blockDim.x + threadIdx.x;     // 32768 threads
    if (idx < (I_ + A_) * H_) { int c = idx / H_, h = idx % H_; W1T[(size_t)h * (I_ + A_) + c] = (__bf16)W1[idx]; }
    if (idx < H_ * A_)        { int c = idx / A_, a = idx % A_; W2T[(size_t)a * H_ + c] = (__bf16)W2[idx]; }
    for (size_t i = idx; i < (size_t)B_ * D_ * A_; i += (size_t)gridDim.x * blockDim.x)
        att[i] = 0.0f;
}

// ----------------------- G = dendron . axon^T / d --------------------------
// grid: x = 72 n-tiles * 18 m-groups(64), y = B; 1 wave/block.
__global__ void kG(const __bf16* __restrict__ dendron, const __bf16* __restrict__ axon,
                   __bf16* __restrict__ G) {
    int w = blockIdx.x, b = blockIdx.y, lane = threadIdx.x;
    int nt = w / 18, mg = w % 18;
    int half = lane >> 4, l16 = lane & 15;
    int n0 = nt * 16, m0 = mg * 64;
    int nrow = n0 + l16;
    v8f acc[4] = {};
    const __bf16* drow = dendron + ((size_t)b * D_ + nrow) * E_;
    for (int e0 = 0; e0 < E_; e0 += 32) {
        v16bf aOp;
        v8bf alo = *(const v8bf*)(drow + e0 + 8 * half);
        v8bf ahi = *(const v8bf*)(drow + e0 + 8 * half + 16);
#pragma unroll
        for (int i = 0; i < 8; ++i) { aOp[i] = alo[i]; aOp[8 + i] = ahi[i]; }
#pragma unroll
        for (int t = 0; t < 4; ++t) {
            int mcol = m0 + t * 16 + l16;
            const __bf16* arow = axon + ((size_t)b * D_ + mcol) * E_ + e0 + 8 * half;
            v16bf bOp;
            v8bf blo = *(const v8bf*)(arow);
            v8bf bhi = *(const v8bf*)(arow + 16);
#pragma unroll
            for (int i = 0; i < 8; ++i) { bOp[i] = blo[i]; bOp[8 + i] = bhi[i]; }
            acc[t] = __builtin_amdgcn_wmma_f32_16x16x32_bf16(
                false, aOp, false, bOp, (short)0, acc[t], false, false);
        }
    }
    const float invd = 1.0f / (float)D_;
#pragma unroll
    for (int t = 0; t < 4; ++t)
#pragma unroll
        for (int v = 0; v < 8; ++v) {
            int row = n0 + v + 8 * half;
            int col = m0 + t * 16 + l16;
            G[((size_t)b * D_ + row) * D_ + col] = (__bf16)(acc[t][v] * invd);
        }
}

// ------------------- per-step prologue (VALU, tiny) ------------------------
__global__ void kPre(const float* __restrict__ att, const __bf16* __restrict__ axon,
                     const float* __restrict__ CEout, const int* __restrict__ program_arg,
                     int l, float* __restrict__ d_s, float* __restrict__ argdot,
                     __bf16* __restrict__ iaT) {
    int idx = blockIdx.x * blockDim.x + threadIdx.x;   // B*D
    if (idx >= B_ * D_) return;
    int b = idx / D_, m = idx % D_;
    const float* arow = att + (size_t)idx * A_;
    float sum = 0.0f;
    for (int a = 0; a < A_; ++a) {
        float v = arow[a];
        sum += v;
        iaT[((size_t)b * (I_ + A_) + I_ + a) * D_ + m] = (__bf16)v;
    }
    d_s[idx] = sum * (1.0f / (float)A_);
    int arg = program_arg[b * L_ + l];
    const float* ce = CEout + (size_t)arg * E_;
    const __bf16* ax = axon + (size_t)idx * E_;
    float dot = 0.0f;
    for (int e = 0; e < E_; ++e) dot += ce[e] * (float)ax[e];
    argdot[idx] = dot * (1.0f / (float)E_);
}

// ------------- mt = clip(G * s) @ [identity|att] / d (fused) ---------------
// grid: x = 18 (groups of 4 n-tiles), y = B; 128 threads = 4 waves.
// iaT K-slab (32 x 128 bf16 = 8KB) double-buffered in LDS, staged with
// async-to-LDS loads (ASYNCcnt) when available; B-tiles read via ds_load_b128.
__global__ void __launch_bounds__(128) kMT(const __bf16* __restrict__ G,
                                           const float* __restrict__ d_s,
                                           const __bf16* __restrict__ iaT,
                                           float* __restrict__ mt) {
    __shared__ __align__(16) __bf16 slab[2][(I_ + A_) * 32];   // [buf][k][m]
    int b = blockIdx.y;
    int tid = threadIdx.x;
    int wave = tid >> 5, lane = tid & 31;
    int half = lane >> 4, l16 = lane & 15;
    int nt = blockIdx.x * 4 + wave;
    int n0 = nt * 16, nrow = n0 + l16;
    const __bf16* grow = G + ((size_t)b * D_ + nrow) * D_;
    const float*  sb   = d_s + (size_t)b * D_;
    const __bf16* srcRow = iaT + ((size_t)b * (I_ + A_) + tid) * D_;  // tid = k row
    v8f acc[8] = {};

    auto stage = [&](int buf, int m0) {
        const __bf16* src = srcRow + m0;              // 32 bf16 = 64B per thread
        __bf16* dst = &slab[buf][tid * 32];
#if USE_ASYNC_LDS
        __builtin_amdgcn_global_load_async_to_lds_b128(
            (GAS_ v4i*)(src),      (LAS_ v4i*)(dst),      0, 0);
        __builtin_amdgcn_global_load_async_to_lds_b128(
            (GAS_ v4i*)(src + 8),  (LAS_ v4i*)(dst + 8),  0, 0);
        __builtin_amdgcn_global_load_async_to_lds_b128(
            (GAS_ v4i*)(src + 16), (LAS_ v4i*)(dst + 16), 0, 0);
        __builtin_amdgcn_global_load_async_to_lds_b128(
            (GAS_ v4i*)(src + 24), (LAS_ v4i*)(dst + 24), 0, 0);
#else
        v8bf x0 = *(const v8bf*)(src);
        v8bf x1 = *(const v8bf*)(src + 8);
        v8bf x2 = *(const v8bf*)(src + 16);
        v8bf x3 = *(const v8bf*)(src + 24);
        *(v8bf*)(dst)      = x0;
        *(v8bf*)(dst + 8)  = x1;
        *(v8bf*)(dst + 16) = x2;
        *(v8bf*)(dst + 24) = x3;
#endif
    };

    stage(0, 0);
    for (int it = 0; it < D_ / 32; ++it) {
        int m0 = it * 32;
#if USE_ASYNC_LDS
        __builtin_amdgcn_s_wait_asynccnt(0);   // my async stores landed in LDS
#endif
        __syncthreads();                       // publish slab to all 4 waves
        if (it + 1 < D_ / 32) stage((it + 1) & 1, m0 + 32);

        // A operand: clip(G*s) -> bf16 (per-lane K runs [kb,kb+8),[kb+16,kb+24))
        int mlo = m0 + 8 * half;
        __builtin_prefetch((const void*)(grow + mlo + 256), 0, 0);   // global_prefetch_b8
        v8bf g0 = *(const v8bf*)(grow + mlo);
        v8bf g1 = *(const v8bf*)(grow + mlo + 16);
        v8f  s0 = *(const v8f*)(sb + mlo);
        v8f  s1 = *(const v8f*)(sb + mlo + 16);
        v16bf aOp;
#pragma unroll
        for (int i = 0; i < 8; ++i) {
            aOp[i]     = (__bf16)clip2((float)g0[i] * s0[i]);
            aOp[8 + i] = (__bf16)clip2((float)g1[i] * s1[i]);
        }
        const __bf16* slb = &slab[it & 1][0];
#pragma unroll
        for (int kt = 0; kt < 8; ++kt) {
            int kcol = kt * 16 + l16;
            const __bf16* srow2 = slb + kcol * 32 + 8 * half;   // ds_load_b128 x2
            v16bf bOp;
            v8bf blo = *(const v8bf*)(srow2);
            v8bf bhi = *(const v8bf*)(srow2 + 16);
#pragma unroll
            for (int i = 0; i < 8; ++i) { bOp[i] = blo[i]; bOp[8 + i] = bhi[i]; }
            acc[kt] = __builtin_amdgcn_wmma_f32_16x16x32_bf16(
                false, aOp, false, bOp, (short)0, acc[kt], false, false);
        }
    }
    const float invd = 1.0f / (float)D_;
#pragma unroll
    for (int kt = 0; kt < 8; ++kt)
#pragma unroll
        for (int v = 0; v < 8; ++v) {
            int row = n0 + v + 8 * half;
            int col = kt * 16 + l16;
            mt[((size_t)b * D_ + row) * (I_ + A_) + col] = acc[kt][v] * invd;
        }
}

// ------------------- hid = relu(mt @ W1 + b1), bf16 ------------------------
__global__ void kMLP1(const float* __restrict__ mt, const __bf16* __restrict__ W1T,
                      const float* __restrict__ b1, __bf16* __restrict__ hid) {
    int rt = blockIdx.x, hb = blockIdx.y, lane = threadIdx.x;
    int half = lane >> 4, l16 = lane & 15;
    int r0 = rt * 16;
    const float* mrow = mt + (size_t)(r0 + l16) * (I_ + A_);
    v8f acc[8] = {};
    for (int c0 = 0; c0 < (I_ + A_); c0 += 32) {
        int clo = c0 + 8 * half;
        v8f x0 = *(const v8f*)(mrow + clo);
        v8f x1 = *(const v8f*)(mrow + clo + 16);
        v16bf aOp;
#pragma unroll
        for (int i = 0; i < 8; ++i) { aOp[i] = (__bf16)x0[i]; aOp[8 + i] = (__bf16)x1[i]; }
        int crow = c0 + 8 * half;
#pragma unroll
        for (int ht = 0; ht < 8; ++ht) {
            int hcol = hb * 128 + ht * 16 + l16;
            const __bf16* wrow = W1T + (size_t)hcol * (I_ + A_) + crow;
            v16bf bOp;
            v8bf blo = *(const v8bf*)(wrow);
            v8bf bhi = *(const v8bf*)(wrow + 16);
#pragma unroll
            for (int i = 0; i < 8; ++i) { bOp[i] = blo[i]; bOp[8 + i] = bhi[i]; }
            acc[ht] = __builtin_amdgcn_wmma_f32_16x16x32_bf16(
                false, aOp, false, bOp, (short)0, acc[ht], false, false);
        }
    }
#pragma unroll
    for (int ht = 0; ht < 8; ++ht)
#pragma unroll
        for (int v = 0; v < 8; ++v) {
            int row = r0 + v + 8 * half;
            int col = hb * 128 + ht * 16 + l16;
            hid[(size_t)row * H_ + col] = (__bf16)fmaxf(acc[ht][v] + b1[col], 0.0f);
        }
}

// --- att_tr = hid@W2 + b2 + mt[:,64:], att = relu(att + ins*att_ins + tra*att_tr)
__global__ void kMLP2(const __bf16* __restrict__ hid, const __bf16* __restrict__ W2T,
                      const float* __restrict__ b2, const float* __restrict__ mt,
                      const float* __restrict__ argdot, const float* __restrict__ meta,
                      const int* __restrict__ program_op, int l,
                      float* __restrict__ att, float* __restrict__ hist) {
    int rt = blockIdx.x, lane = threadIdx.x;
    int half = lane >> 4, l16 = lane & 15;
    int r0 = rt * 16;
    const __bf16* hrow = hid + (size_t)(r0 + l16) * H_;
    v8f acc[4] = {};
    for (int c0 = 0; c0 < H_; c0 += 32) {
        int clo = c0 + 8 * half;
        v8bf h0 = *(const v8bf*)(hrow + clo);
        v8bf h1 = *(const v8bf*)(hrow + clo + 16);
        v16bf aOp;
#pragma unroll
        for (int i = 0; i < 8; ++i) { aOp[i] = h0[i]; aOp[8 + i] = h1[i]; }
        int crow = c0 + 8 * half;
#pragma unroll
        for (int at = 0; at < 4; ++at) {
            int acol = at * 16 + l16;
            const __bf16* wrow = W2T + (size_t)acol * H_ + crow;
            v16bf bOp;
            v8bf blo = *(const v8bf*)(wrow);
            v8bf bhi = *(const v8bf*)(wrow + 16);
#pragma unroll
            for (int i = 0; i < 8; ++i) { bOp[i] = blo[i]; bOp[8 + i] = bhi[i]; }
            acc[at] = __builtin_amdgcn_wmma_f32_16x16x32_bf16(
                false, aOp, false, bOp, (short)0, acc[at], false, false);
        }
    }
#pragma unroll
    for (int at = 0; at < 4; ++at)
#pragma unroll
        for (int v = 0; v < 8; ++v) {
            int row = r0 + v + 8 * half;     // flat over B*D
            int a = at * 16 + l16;
            int b = row / D_, m = row % D_;
            float att_tr = acc[at][v] + b2[a] + mt[(size_t)row * (I_ + A_) + I_ + a];
            int op = program_op[b * L_ + l];
            float ins = (op == 0) ? 1.0f : 0.0f;
            float tra = (op == 1) ? 1.0f : 0.0f;
            float val = att[(size_t)row * A_ + a] + ins * meta[a] * argdot[row] + tra * att_tr;
            val = fmaxf(val, 0.0f);
            att[(size_t)row * A_ + a] = val;
            hist[(((size_t)l * B_ + b) * D_ + m) * A_ + a] = val;
        }
}

// ------------- out = log_softmax(mean(att,-1), axis over D) ----------------
__global__ void kFinal(const float* __restrict__ att, float* __restrict__ out) {
    int b = blockIdx.x, t = threadIdx.x;
    __shared__ float red[256];
    float mx = -1e30f;
    for (int m = t; m < D_; m += 256) {
        const float* row = att + ((size_t)b * D_ + m) * A_;
        float sum = 0.0f;
        for (int a = 0; a < A_; ++a) sum += row[a];
        mx = fmaxf(mx, sum * (1.0f / (float)A_));
    }
    red[t] = mx; __syncthreads();
    for (int o = 128; o > 0; o >>= 1) { if (t < o) red[t] = fmaxf(red[t], red[t + o]); __syncthreads(); }
    mx = red[0]; __syncthreads();
    float se = 0.0f;
    for (int m = t; m < D_; m += 256) {
        const float* row = att + ((size_t)b * D_ + m) * A_;
        float sum = 0.0f;
        for (int a = 0; a < A_; ++a) sum += row[a];
        se += expf(sum * (1.0f / (float)A_) - mx);
    }
    red[t] = se; __syncthreads();
    for (int o = 128; o > 0; o >>= 1) { if (t < o) red[t] += red[t + o]; __syncthreads(); }
    float lse = logf(red[0]) + mx;
    for (int m = t; m < D_; m += 256) {
        const float* row = att + ((size_t)b * D_ + m) * A_;
        float sum = 0.0f;
        for (int a = 0; a < A_; ++a) sum += row[a];
        out[b * D_ + m] = sum * (1.0f / (float)A_) - lse;
    }
}

// ---------------------------------------------------------------------------

extern "C" void kernel_launch(void* const* d_in, const int* in_sizes, int n_in,
                              void* d_out, int out_size, void* d_ws, size_t ws_size,
                              hipStream_t stream) {
    const float* scene       = (const float*)d_in[0];
    const int*   program_op  = (const int*)d_in[1];
    const int*   program_arg = (const int*)d_in[2];
    const float* WfIn  = (const float*)d_in[3];
    const float* bfIn  = (const float*)d_in[4];
    const float* WfOut = (const float*)d_in[5];
    const float* bfOut = (const float*)d_in[6];
    const float* WfId  = (const float*)d_in[7];
    const float* bfId  = (const float*)d_in[8];
    const float* CEin  = (const float*)d_in[9];
    const float* CEout = (const float*)d_in[10];
    const float* CEid  = (const float*)d_in[11];
    const float* meta  = (const float*)d_in[14];   // attention_init
    const float* W1    = (const float*)d_in[15];
    const float* b1    = (const float*)d_in[16];
    const float* W2    = (const float*)d_in[17];
    const float* b2    = (const float*)d_in[18];

    char* ws = (char*)d_ws;
    size_t off = 0;
    auto alloc = [&](size_t bytes) -> void* {
        void* p = ws + off;
        off += (bytes + 255) & ~(size_t)255;
        return p;
    };
    __bf16* dendron = (__bf16*)alloc((size_t)B_ * D_ * E_ * 2);
    __bf16* axon    = (__bf16*)alloc((size_t)B_ * D_ * E_ * 2);
    __bf16* G       = (__bf16*)alloc((size_t)B_ * D_ * D_ * 2);
    __bf16* iaT     = (__bf16*)alloc((size_t)B_ * (I_ + A_) * D_ * 2);
    float*  att     = (float*) alloc((size_t)B_ * D_ * A_ * 4);
    float*  mtbuf   = (float*) alloc((size_t)B_ * D_ * (I_ + A_) * 4);
    __bf16* hid     = (__bf16*)alloc((size_t)B_ * D_ * H_ * 2);
    float*  d_s     = (float*) alloc((size_t)B_ * D_ * 4);
    float*  argdot  = (float*) alloc((size_t)B_ * D_ * 4);
    __bf16* W1T     = (__bf16*)alloc((size_t)H_ * (I_ + A_) * 2);
    __bf16* W2T     = (__bf16*)alloc((size_t)A_ * H_ * 2);
    (void)ws_size; (void)in_sizes; (void)n_in; (void)out_size;

    float* out  = (float*)d_out;
    float* hist = out + (size_t)B_ * D_;

    kConcepts<<<(B_ * M_ + 255) / 256, 256, 0, stream>>>(CEin, CEout, CEid, dendron, axon, iaT);
    kProj<<<(B_ * N_ * (2 * E_ + I_) + 255) / 256, 256, 0, stream>>>(
        scene, WfIn, bfIn, WfOut, bfOut, WfId, bfId, dendron, axon, iaT);
    kWT<<<128, 256, 0, stream>>>(W1, W2, W1T, W2T, att);
    kG<<<dim3(72 * 18, B_), 32, 0, stream>>>(dendron, axon, G);

    for (int l = 0; l < L_; ++l) {
        kPre<<<(B_ * D_ + 255) / 256, 256, 0, stream>>>(att, axon, CEout, program_arg, l,
                                                        d_s, argdot, iaT);
        kMT<<<dim3(D_ / 16 / 4, B_), 128, 0, stream>>>(G, d_s, iaT, mtbuf);
        kMLP1<<<dim3(B_ * D_ / 16, 2), 32, 0, stream>>>(mtbuf, W1T, b1, hid);
        kMLP2<<<B_ * D_ / 16, 32, 0, stream>>>(hid, W2T, b2, mtbuf, argdot, meta,
                                               program_op, l, att, hist);
    }
    kFinal<<<B_, 256, 0, stream>>>(att, out);
}
